// PyG_PointNet2_Alpha_Predictor_78580721647874
// MI455X (gfx1250) — matrix operand order repaired and checked by reading, not compiled, not run on hardware
//
#include <hip/hip_runtime.h>
#include <hip/hip_bf16.h>

typedef __attribute__((ext_vector_type(16))) _Float16 v16h;
typedef __attribute__((ext_vector_type(8)))  _Float16 v8h;
typedef __attribute__((ext_vector_type(8)))  float    v8f;

// Flip to 0 if the assembler rejects the async-to-LDS mnemonic.
#define USE_ASYNC_STAGE 1

// ---------------------------------------------------------------------------
// Zero-padded f32 -> f16 weight conversion: Wh[Kp x N], rows >= Kreal are 0.
// ---------------------------------------------------------------------------
__global__ void convert_w(const float* __restrict__ W, _Float16* __restrict__ Wh,
                          int Kreal, int Kp, int N)
{
    const int tid = blockIdx.x * blockDim.x + threadIdx.x;
    if (tid >= Kp * N) return;
    const int k = tid / N, n = tid % N;
    Wh[tid] = (_Float16)((k < Kreal) ? W[(size_t)k * N + n] : 0.0f);
}

// ---------------------------------------------------------------------------
// GEMM with fused bias + ReLU, f16 in / f16 out, f32 accumulate.
//   C[M,N] = relu(A[M,Kp] * Wh[Kp,N] + bias[N])
// A f16 (zero-padded cols), Wh f16 pre-padded.  Kp%32==0, N%64==0, M%128==0.
// Block = 256 threads = 8 waves.  Each wave computes a 16(M) x 64(N) tile
// (4 WMMA accumulators, A-fragment reused 4x).  The 8 waves share one
// 32(K) x 64(N) weight tile staged through LDS (4 KB) per K-step.
// ---------------------------------------------------------------------------
__global__ void __launch_bounds__(256)
gemm_relu_wmma(const _Float16* __restrict__ A, int Kp,
               const _Float16* __restrict__ Wh,
               const float* __restrict__ bias,
               _Float16* __restrict__ C, int N)
{
    __shared__ _Float16 lw[32 * 64];            // K=32 x N=64 tile
    const int lane = threadIdx.x & 31;
    const int wid  = threadIdx.x >> 5;
    const int m0 = (blockIdx.x * 8 + wid) * 16;
    const int n0 = blockIdx.y * 64;

    // A-fragment addressing per CDNA5 16-bit A 16x32 layout:
    // lanes 0-15: row = lane,    K = {0..7} U {16..23}
    // lanes16-31: row = lane-16, K = {8..15} U {24..31}
    const int rowA = m0 + (lane & 15);
    const int kb   = (lane < 16) ? 0 : 8;
    const _Float16* aRow = A + (size_t)rowA * Kp + kb;

    // Per-thread 16-byte chunk of the 32x64 weight tile.
    const int sk = threadIdx.x >> 3;            // K row 0..31
    const int sc = (threadIdx.x & 7) * 8;       // N col 0,8,...,56
    const _Float16* wSrc = Wh + (size_t)sk * N + n0 + sc;
    _Float16* wDst = &lw[sk * 64 + sc];

    v8f acc[4] = {{}, {}, {}, {}};

    for (int k0 = 0; k0 < Kp; k0 += 32) {
        __syncthreads();
#if USE_ASYNC_STAGE
        // CDNA5 async copy: global -> LDS without touching VGPRs (ASYNCcnt).
        asm volatile("global_load_async_to_lds_b128 %0, %1, off"
                     :: "v"((unsigned)(size_t)wDst),
                        "v"((unsigned long long)(size_t)(wSrc + (size_t)k0 * N))
                     : "memory");
        asm volatile("s_wait_asynccnt 0" ::: "memory");
#else
        *(v8h*)wDst = *(const v8h*)(wSrc + (size_t)k0 * N);
#endif
        __syncthreads();

        const v8h alo = *(const v8h*)(aRow + k0);        // K = kb .. kb+7
        const v8h ahi = *(const v8h*)(aRow + k0 + 16);   // K = kb+16 .. kb+23
        if (k0 + 32 < Kp) {
            __builtin_prefetch(aRow + k0 + 32, 0, 1);    // global_prefetch_b8
        }
        v16h a;
        #pragma unroll
        for (int i = 0; i < 8; ++i) { a[i] = alo[i]; a[8 + i] = ahi[i]; }

        #pragma unroll
        for (int t = 0; t < 4; ++t) {
            // lane l holds B column n = l&15 of sub-tile t; K = kb+{0..7} U kb+16+{0..7}
            v16h b;
            #pragma unroll
            for (int i = 0; i < 8; ++i) {
                b[i]     = lw[(kb + i) * 64 + t * 16 + (lane & 15)];
                b[8 + i] = lw[(kb + 16 + i) * 64 + t * 16 + (lane & 15)];
            }
            acc[t] = __builtin_amdgcn_wmma_f32_16x16x32_f16(
                         false, a, false, b, (short)0, acc[t], false, false);
        }
    }

    // C/D layout: VGPR r -> M = m0 + r (+8 for lanes 16-31), N = lane & 15
    const int nn    = lane & 15;
    const int mbase = m0 + ((lane < 16) ? 0 : 8);
    #pragma unroll
    for (int t = 0; t < 4; ++t) {
        const int n  = n0 + t * 16 + nn;
        const float bb = bias[n];
        #pragma unroll
        for (int r = 0; r < 8; ++r) {
            float v = acc[t][r] + bb;
            v = v > 0.0f ? v : 0.0f;
            C[(size_t)(mbase + r) * N + n] = (_Float16)v;
        }
    }
}

// ---------------------------------------------------------------------------
// Build per-edge feature rows: [ feat[nbr[e]] (wf8*8 halves) | rel-pos (3) | 0s ]
// ---------------------------------------------------------------------------
__global__ void build_edges(_Float16* __restrict__ dst, int ld8,
                            const _Float16* __restrict__ feat, int wf8,
                            const int* __restrict__ nbr,
                            const int* __restrict__ cidx,
                            const float* __restrict__ posl, int nedges)
{
    const int tid = blockIdx.x * blockDim.x + threadIdx.x;
    const int e  = tid / ld8;
    const int c8 = tid % ld8;
    if (e >= nedges) return;
    v8h out = {};
    if (c8 < wf8) {
        const int j = nbr[e];
        out = *(const v8h*)(feat + (size_t)j * (wf8 * 8) + c8 * 8);
    } else if (c8 == wf8) {
        const int j = nbr[e];
        const int c = cidx[e >> 6];
        out[0] = (_Float16)(posl[3 * j + 0] - posl[3 * c + 0]);
        out[1] = (_Float16)(posl[3 * j + 1] - posl[3 * c + 1]);
        out[2] = (_Float16)(posl[3 * j + 2] - posl[3 * c + 2]);
    }
    *(v8h*)(dst + (size_t)e * (ld8 * 8) + c8 * 8) = out;
}

// Max over groups of 64 consecutive rows: X[m, c] = max_k H[m*64+k, c]
__global__ void segmax64(const _Float16* __restrict__ H,
                         _Float16* __restrict__ X, int C, int M)
{
    const int tid = blockIdx.x * blockDim.x + threadIdx.x;
    const int m = tid / C;
    const int c = tid % C;
    if (m >= M) return;
    const _Float16* p = H + (size_t)m * 64 * C + c;
    float v = -3.4e38f;
    for (int k = 0; k < 64; ++k) {
        const float t = (float)p[(size_t)k * C];
        v = t > v ? t : v;
    }
    X[(size_t)m * C + c] = (_Float16)v;
}

// dst[row] = [ A[row/divA] (wA8*8) | B[row] (rest) ]   (nearest upsample concat)
__global__ void concat_up(_Float16* __restrict__ dst, int ld8,
                          const _Float16* __restrict__ A, int wA8, int divA,
                          const _Float16* __restrict__ B, int rows)
{
    const int tid = blockIdx.x * blockDim.x + threadIdx.x;
    const int row = tid / ld8;
    const int c8  = tid % ld8;
    if (row >= rows) return;
    const int wB8 = ld8 - wA8;
    v8h o;
    if (c8 < wA8)
        o = *(const v8h*)(A + (size_t)(row / divA) * (wA8 * 8) + c8 * 8);
    else
        o = *(const v8h*)(B + (size_t)row * (wB8 * 8) + (c8 - wA8) * 8);
    *(v8h*)(dst + (size_t)row * (ld8 * 8) + c8 * 8) = o;
}

// dst[row] (ld=160) = [ A[row/4] (128 halves) | pos[row] (3 f32->f16) | 0s ]
__global__ void concat_up_pos(_Float16* __restrict__ dst,
                              const _Float16* __restrict__ A,
                              const float* __restrict__ pos, int rows)
{
    const int ld8 = 20, wA8 = 16;
    const int tid = blockIdx.x * blockDim.x + threadIdx.x;
    const int row = tid / ld8;
    const int c8  = tid % ld8;
    if (row >= rows) return;
    v8h o = {};
    if (c8 < wA8) {
        o = *(const v8h*)(A + (size_t)(row >> 2) * 128 + c8 * 8);
    } else if (c8 == wA8) {
        o[0] = (_Float16)pos[3 * row + 0];
        o[1] = (_Float16)pos[3 * row + 1];
        o[2] = (_Float16)pos[3 * row + 2];
    }
    *(v8h*)(dst + (size_t)row * 160 + c8 * 8) = o;
}

__global__ void gather_pos(float* __restrict__ dst, const float* __restrict__ src,
                           const int* __restrict__ idx, int n)
{
    const int i = blockIdx.x * blockDim.x + threadIdx.x;
    if (i >= n) return;
    const int j = idx[i];
    dst[3 * i + 0] = src[3 * j + 0];
    dst[3 * i + 1] = src[3 * j + 1];
    dst[3 * i + 2] = src[3 * j + 2];
}

// out[i] = softplus( H[i,:] . W2 + b2 ),  H = relu(head1) f16 [n,64]
__global__ void head2_softplus(const _Float16* __restrict__ H,
                               const float* __restrict__ W2,
                               const float* __restrict__ b2,
                               float* __restrict__ out, int n)
{
    const int i = blockIdx.x * blockDim.x + threadIdx.x;
    if (i >= n) return;
    float acc = b2[0];
    const _Float16* h = H + (size_t)i * 64;
    #pragma unroll 8
    for (int c = 0; c < 64; ++c) acc += (float)h[c] * W2[c];
    out[i] = acc > 0.0f ? acc + log1pf(expf(-acc)) : log1pf(expf(acc));
}

// ---------------------------------------------------------------------------
static inline void gemm(const _Float16* A, int Kp, const _Float16* Wh,
                        const float* b, _Float16* C, int M, int N,
                        hipStream_t s)
{
    dim3 grid(M / 128, N / 64);
    gemm_relu_wmma<<<grid, 256, 0, s>>>(A, Kp, Wh, b, C, N);
}

extern "C" void kernel_launch(void* const* d_in, const int* in_sizes, int n_in,
                              void* d_out, int out_size, void* d_ws, size_t ws_size,
                              hipStream_t stream) {
    (void)in_sizes; (void)n_in; (void)out_size; (void)ws_size;

    // ---- inputs (setup_inputs dict order, params flattened (W,b) pairs) ----
    const float* pos  = (const float*)d_in[0];
    const float* m1W0 = (const float*)d_in[1];  const float* m1b0 = (const float*)d_in[2];
    const float* m1W1 = (const float*)d_in[3];  const float* m1b1 = (const float*)d_in[4];
    const float* m1W2 = (const float*)d_in[5];  const float* m1b2 = (const float*)d_in[6];
    const float* m2W0 = (const float*)d_in[7];  const float* m2b0 = (const float*)d_in[8];
    const float* m2W1 = (const float*)d_in[9];  const float* m2b1 = (const float*)d_in[10];
    const float* m2W2 = (const float*)d_in[11]; const float* m2b2 = (const float*)d_in[12];
    const float* m3W0 = (const float*)d_in[13]; const float* m3b0 = (const float*)d_in[14];
    const float* m3W1 = (const float*)d_in[15]; const float* m3b1 = (const float*)d_in[16];
    const float* m3W2 = (const float*)d_in[17]; const float* m3b2 = (const float*)d_in[18];
    const float* f3W0 = (const float*)d_in[19]; const float* f3b0 = (const float*)d_in[20];
    const float* f3W1 = (const float*)d_in[21]; const float* f3b1 = (const float*)d_in[22];
    const float* f2W0 = (const float*)d_in[23]; const float* f2b0 = (const float*)d_in[24];
    const float* f2W1 = (const float*)d_in[25]; const float* f2b1 = (const float*)d_in[26];
    const float* f1W0 = (const float*)d_in[27]; const float* f1b0 = (const float*)d_in[28];
    const float* f1W1 = (const float*)d_in[29]; const float* f1b1 = (const float*)d_in[30];
    const float* h1W  = (const float*)d_in[31]; const float* h1b  = (const float*)d_in[32];
    const float* h2W  = (const float*)d_in[33]; const float* h2b  = (const float*)d_in[34];
    const int* nbr1 = (const int*)d_in[35]; const int* idx1 = (const int*)d_in[36];
    const int* nbr2 = (const int*)d_in[37]; const int* idx2 = (const int*)d_in[38];
    const int* nbr3 = (const int*)d_in[39]; const int* idx3 = (const int*)d_in[40];

    // ---- workspace carving (all offsets 16B-aligned; sizes padded) ----
    char* w = (char*)d_ws;
    _Float16* E    = (_Float16*)w; w += (size_t)10485760 * 2;  // edges (max 65536x160)
    _Float16* HA   = (_Float16*)w; w += (size_t)16777216 * 2;  // ping (max 16384x1024)
    _Float16* HB   = (_Float16*)w; w += (size_t)8388608  * 2;  // pong (max 65536x128)
    _Float16* x1   = (_Float16*)w; w += (size_t)4096 * 128 * 2;
    _Float16* x2   = (_Float16*)w; w += (size_t)1024 * 256 * 2;
    _Float16* x3   = (_Float16*)w; w += (size_t)256 * 1024 * 2;
    _Float16* x2fp = (_Float16*)w; w += (size_t)1024 * 256 * 2;
    _Float16* x1fp = (_Float16*)w; w += (size_t)4096 * 128 * 2;
    _Float16* x0fp = (_Float16*)w; w += (size_t)16384 * 128 * 2;
    float*    pos1 = (float*)w;    w += (size_t)4096 * 3 * 4;
    float*    pos2 = (float*)w;    w += (size_t)1024 * 3 * 4;
    _Float16* wpool = (_Float16*)((size_t)(w + 255) & ~(size_t)255);

    // ---- pre-convert all weights to zero-padded f16 [Kp x N] ----
    _Float16* wp = wpool;
    auto conv = [&](const float* W, int Kreal, int Kp, int N) -> _Float16* {
        _Float16* dst = wp; wp += (size_t)Kp * N;
        convert_w<<<(Kp * N + 255) / 256, 256, 0, stream>>>(W, dst, Kreal, Kp, N);
        return dst;
    };
    _Float16* m1W0h = conv(m1W0,    3,   32,   64);
    _Float16* m1W1h = conv(m1W1,   64,   64,   64);
    _Float16* m1W2h = conv(m1W2,   64,   64,  128);
    _Float16* m2W0h = conv(m2W0,  131,  160,  128);
    _Float16* m2W1h = conv(m2W1,  128,  128,  128);
    _Float16* m2W2h = conv(m2W2,  128,  128,  256);
    _Float16* m3W0h = conv(m3W0,  259,  288,  256);
    _Float16* m3W1h = conv(m3W1,  256,  256,  512);
    _Float16* m3W2h = conv(m3W2,  512,  512, 1024);
    _Float16* f3W0h = conv(f3W0, 1280, 1280,  256);
    _Float16* f3W1h = conv(f3W1,  256,  256,  256);
    _Float16* f2W0h = conv(f2W0,  384,  384,  128);
    _Float16* f2W1h = conv(f2W1,  128,  128,  128);
    _Float16* f1W0h = conv(f1W0,  131,  160,  128);
    _Float16* f1W1h = conv(f1W1,  128,  128,  128);
    _Float16* h1Wh  = conv(h1W,   128,  128,   64);

    gather_pos<<<(4096 + 255) / 256, 256, 0, stream>>>(pos1, pos, idx1, 4096);
    gather_pos<<<(1024 + 255) / 256, 256, 0, stream>>>(pos2, pos1, idx2, 1024);

    // ---- SA1: 4096 queries in 4 chunks of 1024 (65536 edges each) ----
    for (int ch = 0; ch < 4; ++ch) {
        const int q0 = ch * 1024;
        build_edges<<<(65536 * 4 + 255) / 256, 256, 0, stream>>>(
            E, 4, nullptr, 0, nbr1 + (size_t)q0 * 64, idx1 + q0, pos, 65536);
        gemm(E,  32, m1W0h, m1b0, HA, 65536,  64, stream);
        gemm(HA, 64, m1W1h, m1b1, HB, 65536,  64, stream);
        gemm(HB, 64, m1W2h, m1b2, HA, 65536, 128, stream);
        segmax64<<<(1024 * 128 + 255) / 256, 256, 0, stream>>>(
            HA, x1 + (size_t)q0 * 128, 128, 1024);
    }

    // ---- SA2: 1024 queries, 65536 edges, rows [x1(128)|rel(3)|pad] = 160 ----
    build_edges<<<(65536 * 20 + 255) / 256, 256, 0, stream>>>(
        E, 20, x1, 16, nbr2, idx2, pos1, 65536);
    gemm(E,  160, m2W0h, m2b0, HA, 65536, 128, stream);
    gemm(HA, 128, m2W1h, m2b1, HB, 65536, 128, stream);
    gemm(HB, 128, m2W2h, m2b2, HA, 65536, 256, stream);
    segmax64<<<(1024 * 256 + 255) / 256, 256, 0, stream>>>(HA, x2, 256, 1024);

    // ---- SA3: 256 queries, 16384 edges, rows [x2(256)|rel(3)|pad] = 288 ----
    build_edges<<<(16384 * 36 + 255) / 256, 256, 0, stream>>>(
        E, 36, x2, 32, nbr3, idx3, pos2, 16384);
    gemm(E,  288, m3W0h, m3b0, HA, 16384,  256, stream);
    gemm(HA, 256, m3W1h, m3b1, HB, 16384,  512, stream);
    gemm(HB, 512, m3W2h, m3b2, HA, 16384, 1024, stream);
    segmax64<<<(256 * 1024 + 255) / 256, 256, 0, stream>>>(HA, x3, 1024, 256);

    // ---- FP3: rows [up4(x3)(1024) | x2(256)] = 1280 ----
    concat_up<<<(1024 * 160 + 255) / 256, 256, 0, stream>>>(E, 160, x3, 128, 4, x2, 1024);
    gemm(E,  1280, f3W0h, f3b0, HA,   1024, 256, stream);
    gemm(HA,  256, f3W1h, f3b1, x2fp, 1024, 256, stream);

    // ---- FP2: rows [up4(x2fp)(256) | x1(128)] = 384 ----
    concat_up<<<(4096 * 48 + 255) / 256, 256, 0, stream>>>(E, 48, x2fp, 32, 4, x1, 4096);
    gemm(E,  384, f2W0h, f2b0, HA,   4096, 128, stream);
    gemm(HA, 128, f2W1h, f2b1, x1fp, 4096, 128, stream);

    // ---- FP1: rows [up4(x1fp)(128) | pos(3) | pad] = 160 ----
    concat_up_pos<<<(16384 * 20 + 255) / 256, 256, 0, stream>>>(E, x1fp, pos, 16384);
    gemm(E,  160, f1W0h, f1b0, HA,   16384, 128, stream);
    gemm(HA, 128, f1W1h, f1b1, x0fp, 16384, 128, stream);

    // ---- head: relu(x0fp @ W1 + b1) @ W2 + b2 -> softplus ----
    gemm(x0fp, 128, h1Wh, h1b, HA, 16384, 64, stream);
    head2_softplus<<<(16384 + 255) / 256, 256, 0, stream>>>(
        HA, h2W, h2b, (float*)d_out, 16384);
}
